// MultiheadSelfAttention_73632919323104
// MI455X (gfx1250) — matrix operand order repaired and checked by reading, not compiled, not run on hardware
//
#include <hip/hip_runtime.h>

#define B_  4
#define S_  1024
#define D_  1024
#define H_  16
#define HD_ 64
#define N3_ 3072

typedef __attribute__((ext_vector_type(16))) __bf16 v16bf;
typedef __attribute__((ext_vector_type(8)))  float  v8f;

union Frag {
  v16bf v;
  unsigned short u[16];
  uint4 q[2];
};

__device__ __forceinline__ unsigned short f2bf(float f) {
  unsigned u = __float_as_uint(f);
  u += 0x7FFFu + ((u >> 16) & 1u);   // round-to-nearest-even
  return (unsigned short)(u >> 16);
}

__global__ void mhsa_cast_bf16(const float* __restrict__ src,
                               unsigned short* __restrict__ dst, int n) {
  int i = blockIdx.x * blockDim.x + threadIdx.x;
  if (i < n) dst[i] = f2bf(src[i]);
}

// transpose+cast: src is [K, N] row-major fp32, dst is [N, K] row-major bf16
__global__ void mhsa_cast_tr_bf16(const float* __restrict__ src,
                                  unsigned short* __restrict__ dst,
                                  int K, int N) {
  int i = blockIdx.x * blockDim.x + threadIdx.x;
  if (i < K * N) {
    int n = i / K, k = i % K;
    dst[i] = f2bf(src[(size_t)k * N + n]);
  }
}

// ---------------------------------------------------------------------------
// QKV GEMM: A = xb [4096 x 1024] bf16 row-major, BT = wibT [3072 x 1024] bf16
// (N-major => B fragments are contiguous b128 loads). Each wave computes a
// 32x32 tile (2x2 wmma register block); block = 4 waves = 64x64 tile.
// Epilogue scatters to q/k head-major [B,H,S,HD] and V transposed [B,H,HD,S].
// grid = (3072/64, 4096/64), block = 128
// ---------------------------------------------------------------------------
__global__ void __launch_bounds__(128)
mhsa_gemm_qkv(const unsigned short* __restrict__ xb,
              const unsigned short* __restrict__ wibT,
              const float* __restrict__ b_in,
              unsigned short* __restrict__ qb,
              unsigned short* __restrict__ kbuf,
              unsigned short* __restrict__ vT) {
  const int wave = threadIdx.x >> 5;
  const int lane = threadIdx.x & 31;
  const int half = lane >> 4;
  const int lr   = lane & 15;
  const int mW = blockIdx.y * 64 + (wave >> 1) * 32;
  const int nW = blockIdx.x * 64 + (wave & 1) * 32;

  const unsigned short* arow0 = xb   + (size_t)(mW + lr) * D_;
  const unsigned short* arow1 = xb   + (size_t)(mW + 16 + lr) * D_;
  const unsigned short* brow0 = wibT + (size_t)(nW + lr) * D_;
  const unsigned short* brow1 = wibT + (size_t)(nW + 16 + lr) * D_;

  v8f acc[2][2];
#pragma unroll
  for (int i = 0; i < 2; ++i)
#pragma unroll
    for (int j = 0; j < 2; ++j) acc[i][j] = (v8f){};

  for (int k0 = 0; k0 < D_; k0 += 32) {
    Frag a0, a1, b0, b1;
    a0.q[0] = *(const uint4*)&arow0[k0 + half * 8];
    a0.q[1] = *(const uint4*)&arow0[k0 + 16 + half * 8];
    a1.q[0] = *(const uint4*)&arow1[k0 + half * 8];
    a1.q[1] = *(const uint4*)&arow1[k0 + 16 + half * 8];
    b0.q[0] = *(const uint4*)&brow0[k0 + half * 8];
    b0.q[1] = *(const uint4*)&brow0[k0 + 16 + half * 8];
    b1.q[0] = *(const uint4*)&brow1[k0 + half * 8];
    b1.q[1] = *(const uint4*)&brow1[k0 + 16 + half * 8];
    if (k0 + 32 < D_) {
      __builtin_prefetch(&arow0[k0 + 32], 0, 3);
      __builtin_prefetch(&brow0[k0 + 32], 0, 3);
    }
    acc[0][0] = __builtin_amdgcn_wmma_f32_16x16x32_bf16(false, a0.v, false, b0.v, (short)0, acc[0][0], false, false);
    acc[0][1] = __builtin_amdgcn_wmma_f32_16x16x32_bf16(false, a0.v, false, b1.v, (short)0, acc[0][1], false, false);
    acc[1][0] = __builtin_amdgcn_wmma_f32_16x16x32_bf16(false, a1.v, false, b0.v, (short)0, acc[1][0], false, false);
    acc[1][1] = __builtin_amdgcn_wmma_f32_16x16x32_bf16(false, a1.v, false, b1.v, (short)0, acc[1][1], false, false);
  }

#pragma unroll
  for (int ti = 0; ti < 2; ++ti) {
#pragma unroll
    for (int tj = 0; tj < 2; ++tj) {
#pragma unroll
      for (int j = 0; j < 8; ++j) {
        int m = mW + ti * 16 + half * 8 + j;
        int n = nW + tj * 16 + lr;
        float val = acc[ti][tj][j] + b_in[n];
        int bb = m >> 10, s = m & (S_ - 1);
        int part = n / D_, jj = n % D_;
        int h = jj >> 6, hd = jj & (HD_ - 1);
        size_t headBase = ((size_t)bb * H_ + h);
        if (part == 0)
          qb[(headBase * S_ + s) * HD_ + hd] = f2bf(val * 0.03125f); // 1024^-0.5
        else if (part == 1)
          kbuf[(headBase * S_ + s) * HD_ + hd] = f2bf(val);
        else
          vT[(headBase * HD_ + hd) * S_ + s] = f2bf(val);            // transposed
      }
    }
  }
}

// ---------------------------------------------------------------------------
// Attention: one block per (b, 16-query tile); loops over all 16 heads so the
// head-mean reduces in LDS (no global atomics). Per head: scores 16x1024 fp32
// in LDS via WMMA, parallel softmax (8 threads/row, LDS tree reduction),
// probs packed to bf16 tile, P@V via WMMA, ctx written per head. Finally the
// accumulated mean tile is written once, coalesced.
// LDS: 64KB scores + 64KB mean + 32KB bf16 probs = 160KB (half a WGP).
// grid = (S/16, B), block = 128 (4 waves)
// ---------------------------------------------------------------------------
__global__ void __launch_bounds__(128)
mhsa_attention(const unsigned short* __restrict__ qb,
               const unsigned short* __restrict__ kbuf,
               const unsigned short* __restrict__ vT,
               unsigned short* __restrict__ ctx,
               float* __restrict__ mean_out) {
  __shared__ __align__(16) float          Sc[16 * S_];    // 64 KB scores
  __shared__ __align__(16) float          Macc[16 * S_];  // 64 KB head-mean acc
  __shared__ __align__(16) unsigned short Pb[16 * S_];    // 32 KB bf16 probs
  float* red = (float*)Pb;   // reduction scratch aliased into Pb (dead then)

  const int tid  = threadIdx.x;
  const int wave = tid >> 5;
  const int lane = tid & 31;
  const int half = lane >> 4;
  const int lr   = lane & 15;
  const int qBase = blockIdx.x * 16;
  const int bb    = blockIdx.y;
  const int r  = tid & 15;    // softmax: row owned
  const int sg = tid >> 4;    // softmax: 128-wide segment owned

  for (int i = tid; i < 16 * S_; i += 128) Macc[i] = 0.0f;

  for (int h = 0; h < H_; ++h) {
    __syncthreads();   // Sc/Pb consumers of previous head are done

    const size_t headBase = ((size_t)bb * H_ + h);
    const unsigned short* Qh  = qb   + headBase * S_ * HD_;
    const unsigned short* Kh  = kbuf + headBase * S_ * HD_;
    const unsigned short* VhT = vT   + headBase * HD_ * S_;

    // Q tile as two 16x32 A fragments (hd 0..31 and 32..63)
    Frag qa0, qa1;
    const unsigned short* qrow = Qh + (size_t)(qBase + lr) * HD_;
    qa0.q[0] = *(const uint4*)&qrow[half * 8];
    qa0.q[1] = *(const uint4*)&qrow[16 + half * 8];
    qa1.q[0] = *(const uint4*)&qrow[32 + half * 8];
    qa1.q[1] = *(const uint4*)&qrow[48 + half * 8];

    // scores: wave w covers key blocks w*16, w*16+64, ...
    for (int kb0 = wave * 16; kb0 < S_; kb0 += 64) {
      const unsigned short* kcol = Kh + (size_t)(kb0 + lr) * HD_;
      Frag b0, b1;
      b0.q[0] = *(const uint4*)&kcol[half * 8];
      b0.q[1] = *(const uint4*)&kcol[16 + half * 8];
      b1.q[0] = *(const uint4*)&kcol[32 + half * 8];
      b1.q[1] = *(const uint4*)&kcol[48 + half * 8];

      v8f s = {};
      s = __builtin_amdgcn_wmma_f32_16x16x32_bf16(false, qa0.v, false, b0.v,
                                                  (short)0, s, false, false);
      s = __builtin_amdgcn_wmma_f32_16x16x32_bf16(false, qa1.v, false, b1.v,
                                                  (short)0, s, false, false);
#pragma unroll
      for (int j = 0; j < 8; ++j)
        Sc[(half * 8 + j) * S_ + kb0 + lr] = s[j];
    }
    __syncthreads();

    // ---- parallel softmax: 8 threads per row, 128 elements each ----
    float* row = &Sc[r * S_];
    const int k0 = sg * 128;

    float pm = -3.0e38f;
    for (int k = k0; k < k0 + 128; ++k) pm = fmaxf(pm, row[k]);
    red[r * 8 + sg] = pm;
    __syncthreads();
    if (tid < 16) {
      float m2 = red[tid * 8];
      for (int s2 = 1; s2 < 8; ++s2) m2 = fmaxf(m2, red[tid * 8 + s2]);
      red[128 + tid] = m2;
    }
    __syncthreads();

    const float mx = red[128 + r];
    float ps = 0.0f;
    for (int k = k0; k < k0 + 128; ++k) {
      float e = __expf(row[k] - mx);
      row[k] = e;
      ps += e;
    }
    __syncthreads();           // partial-max slots fully consumed
    red[r * 8 + sg] = ps;
    __syncthreads();
    if (tid < 16) {
      float s2 = 0.0f;
      for (int j = 0; j < 8; ++j) s2 += red[tid * 8 + j];
      red[128 + tid] = 1.0f / s2;
    }
    __syncthreads();
    const float inv = red[128 + r];
    __syncthreads();           // everyone grabbed inv; Pb may now be clobbered

    for (int k = k0; k < k0 + 128; ++k) {
      float p = row[k] * inv;
      Macc[r * S_ + k] += p;           // head-mean accumulation (no atomics)
      Pb[r * S_ + k] = f2bf(p);
    }
    __syncthreads();

    // ---- P @ V : wave w owns hd block w*16 ----
    const int hdB = wave * 16;
    const unsigned short* vTrow = VhT + (size_t)(hdB + lr) * S_;
    const unsigned short* prow  = &Pb[lr * S_];
    v8f acc = {};
    for (int kc = 0; kc < S_; kc += 32) {
      Frag a;   // bf16 prob rows in LDS: contiguous ds_load_b128
      a.q[0] = *(const uint4*)&prow[kc + half * 8];
      a.q[1] = *(const uint4*)&prow[kc + 16 + half * 8];
      Frag b;   // V^T: contiguous key runs per lane
      b.q[0] = *(const uint4*)&vTrow[kc + half * 8];
      b.q[1] = *(const uint4*)&vTrow[kc + 16 + half * 8];
      acc = __builtin_amdgcn_wmma_f32_16x16x32_bf16(
          false, a.v, false, b.v, (short)0, acc, false, false);
    }

    // ctx layout [B, S, H, HD] == row-major [B*S, D] after head concat
#pragma unroll
    for (int j = 0; j < 8; ++j) {
      int q = qBase + half * 8 + j;
      size_t di = (((size_t)bb * S_ + q) * H_ + h) * HD_ + hdB + lr;
      ctx[di] = f2bf(acc[j]);
    }
  }

  // ---- write head-averaged attention, one plain store per element ----
  __syncthreads();
  for (int i = tid; i < 16 * S_; i += 128) {
    int rr = i >> 10, k = i & (S_ - 1);
    mean_out[((size_t)bb * S_ + qBase + rr) * S_ + k] = Macc[i] * (1.0f / H_);
  }
}

// ---------------------------------------------------------------------------
// Output projection: A = ctx [4096 x 1024] bf16, BT = wobT [1024 x 1024] bf16
// (N-major). 32x32 tile per wave, fp32 out + b_out.
// grid = (1024/64, 4096/64), block = 128
// ---------------------------------------------------------------------------
__global__ void __launch_bounds__(128)
mhsa_gemm_out(const unsigned short* __restrict__ ctxb,
              const unsigned short* __restrict__ wobT,
              const float* __restrict__ b_out,
              float* __restrict__ out) {
  const int wave = threadIdx.x >> 5;
  const int lane = threadIdx.x & 31;
  const int half = lane >> 4;
  const int lr   = lane & 15;
  const int mW = blockIdx.y * 64 + (wave >> 1) * 32;
  const int nW = blockIdx.x * 64 + (wave & 1) * 32;

  const unsigned short* arow0 = ctxb + (size_t)(mW + lr) * D_;
  const unsigned short* arow1 = ctxb + (size_t)(mW + 16 + lr) * D_;
  const unsigned short* brow0 = wobT + (size_t)(nW + lr) * D_;
  const unsigned short* brow1 = wobT + (size_t)(nW + 16 + lr) * D_;

  v8f acc[2][2];
#pragma unroll
  for (int i = 0; i < 2; ++i)
#pragma unroll
    for (int j = 0; j < 2; ++j) acc[i][j] = (v8f){};

  for (int k0 = 0; k0 < D_; k0 += 32) {
    Frag a0, a1, b0, b1;
    a0.q[0] = *(const uint4*)&arow0[k0 + half * 8];
    a0.q[1] = *(const uint4*)&arow0[k0 + 16 + half * 8];
    a1.q[0] = *(const uint4*)&arow1[k0 + half * 8];
    a1.q[1] = *(const uint4*)&arow1[k0 + 16 + half * 8];
    b0.q[0] = *(const uint4*)&brow0[k0 + half * 8];
    b0.q[1] = *(const uint4*)&brow0[k0 + 16 + half * 8];
    b1.q[0] = *(const uint4*)&brow1[k0 + half * 8];
    b1.q[1] = *(const uint4*)&brow1[k0 + 16 + half * 8];
    if (k0 + 32 < D_) {
      __builtin_prefetch(&arow0[k0 + 32], 0, 3);
      __builtin_prefetch(&brow0[k0 + 32], 0, 3);
    }
    acc[0][0] = __builtin_amdgcn_wmma_f32_16x16x32_bf16(false, a0.v, false, b0.v, (short)0, acc[0][0], false, false);
    acc[0][1] = __builtin_amdgcn_wmma_f32_16x16x32_bf16(false, a0.v, false, b1.v, (short)0, acc[0][1], false, false);
    acc[1][0] = __builtin_amdgcn_wmma_f32_16x16x32_bf16(false, a1.v, false, b0.v, (short)0, acc[1][0], false, false);
    acc[1][1] = __builtin_amdgcn_wmma_f32_16x16x32_bf16(false, a1.v, false, b1.v, (short)0, acc[1][1], false, false);
  }

#pragma unroll
  for (int ti = 0; ti < 2; ++ti) {
#pragma unroll
    for (int tj = 0; tj < 2; ++tj) {
#pragma unroll
      for (int j = 0; j < 8; ++j) {
        int m = mW + ti * 16 + half * 8 + j;
        int n = nW + tj * 16 + lr;
        out[(size_t)m * D_ + n] = acc[ti][tj][j] + b_out[n];
      }
    }
  }
}

// ---------------------------------------------------------------------------
extern "C" void kernel_launch(void* const* d_in, const int* in_sizes, int n_in,
                              void* d_out, int out_size, void* d_ws, size_t ws_size,
                              hipStream_t stream) {
  const float* x     = (const float*)d_in[0];
  const float* w_in  = (const float*)d_in[1];
  const float* b_in  = (const float*)d_in[2];
  const float* w_out = (const float*)d_in[3];
  const float* b_out = (const float*)d_in[4];

  char* ws = (char*)d_ws;
  unsigned short* xb   = (unsigned short*)(ws);                       // 8 MB
  unsigned short* wibT = (unsigned short*)(ws + ( 8ull << 20));       // 6 MB
  unsigned short* wobT = (unsigned short*)(ws + (14ull << 20));       // 2 MB
  unsigned short* qb   = (unsigned short*)(ws + (16ull << 20));       // 8 MB
  unsigned short* kb   = (unsigned short*)(ws + (24ull << 20));       // 8 MB
  unsigned short* vT   = (unsigned short*)(ws + (32ull << 20));       // 8 MB
  unsigned short* ctx  = (unsigned short*)(ws + (40ull << 20));       // 8 MB

  float* outp  = (float*)d_out;                      // [B,S,D]
  float* meanp = outp + (size_t)B_ * S_ * D_;        // [B,S,S]

  mhsa_cast_bf16<<<(B_ * S_ * D_ + 255) / 256, 256, 0, stream>>>(x, xb, B_ * S_ * D_);
  mhsa_cast_tr_bf16<<<(D_ * N3_ + 255) / 256, 256, 0, stream>>>(w_in, wibT, D_, N3_);
  mhsa_cast_tr_bf16<<<(D_ * D_ + 255) / 256, 256, 0, stream>>>(w_out, wobT, D_, D_);

  mhsa_gemm_qkv<<<dim3(N3_ / 64, (B_ * S_) / 64), 128, 0, stream>>>(
      xb, wibT, b_in, qb, kb, vT);

  mhsa_attention<<<dim3(S_ / 16, B_), 128, 0, stream>>>(qb, kb, vT, ctx, meanp);

  mhsa_gemm_out<<<dim3(D_ / 64, (B_ * S_) / 64), 128, 0, stream>>>(
      ctx, wobT, b_out, outp);
}
